// SwinWindowAttention_8985071583542
// MI455X (gfx1250) — matrix operand order repaired and checked by reading, not compiled, not run on hardware
//
#include <hip/hip_runtime.h>
#include <hip/hip_bf16.h>

// ---------------------------------------------------------------------------
// Swin window attention, fused, BF16 WMMA (gfx1250 / CDNA5, wave32)
// ---------------------------------------------------------------------------

typedef __bf16 v16bf __attribute__((ext_vector_type(16)));
typedef float  v8f   __attribute__((ext_vector_type(8)));

union Frag {
    v16bf v;
    uint4 q[2];
};

__device__ __forceinline__ unsigned short f2bf(float f) {
    union { float f; unsigned u; } v; v.f = f;
    unsigned r = v.u + 0x7FFFu + ((v.u >> 16) & 1u);   // round-to-nearest-even
    return (unsigned short)(r >> 16);
}

// -------------------- constants --------------------
#define NTOK   49          // tokens per window
#define NPAD   64          // padded tokens (4 x 16)
#define CDIM   128         // channels
#define HEADS  4
#define HD     32          // head dim
#define NWIN   64          // mask windows

#define XB_STRIDE 136      // bf16 elems per row of x / attn-out buffer
#define QK_STRIDE 40       // bf16 elems per row of q/k buffers
#define VT_STRIDE 72       // bf16 elems per row of v-transposed buffer
#define AT_STRIDE 72       // f32 elems per row of attention logits

// LDS byte offsets (all 16B aligned)
#define OFF_XB    0                         // 64*136*2      = 17408
#define OFF_QB    17408                     // 4*64*40*2     = 20480
#define OFF_KB    37888                     // 4*64*40*2     = 20480
#define OFF_VT    58368                     // 4*32*72*2     = 18432
#define OFF_BIAS  76800                     // 384*4         = 1536
#define OFF_ATT   78336                     // 4*64*72*4     = 73728
#define LDS_BYTES 152064

// workspace byte offsets
#define WS_WQ   0          // 384*128 bf16  = 98304 B
#define WS_WP   98304      // 128*128 bf16  = 32768 B
#define WS_RPB  131072     // 4*49*49 f32   = 38416 B

// ---------------------------------------------------------------------------
// Prep: f32->bf16 weight conversion + relative-position-bias gather
// ---------------------------------------------------------------------------
__global__ void swin_prep_kernel(const float* __restrict__ qkv_w,
                                 const float* __restrict__ proj_w,
                                 const float* __restrict__ bias_table,
                                 const int*   __restrict__ rel_idx,
                                 unsigned short* __restrict__ wq,
                                 unsigned short* __restrict__ wp,
                                 float* __restrict__ rpb) {
    int i = blockIdx.x * 256 + threadIdx.x;
    if (i < 49152) {
        wq[i] = f2bf(qkv_w[i]);
    } else if (i < 65536) {
        int j = i - 49152;
        wp[j] = f2bf(proj_w[j]);
    } else if (i < 65536 + HEADS * NTOK * NTOK) {
        int j   = i - 65536;
        int h   = j / (NTOK * NTOK);
        int rem = j % (NTOK * NTOK);            // row*49+col
        rpb[j]  = bias_table[rel_idx[rem] * HEADS + h];
    }
}

// ---------------------------------------------------------------------------
// Fused window-attention kernel: one workgroup (8 wave32) per window
// ---------------------------------------------------------------------------
__global__ __launch_bounds__(256)
void swin_attn_kernel(const float* __restrict__ x,
                      const float* __restrict__ mask,
                      const float* __restrict__ qkv_b,
                      const float* __restrict__ proj_b,
                      const unsigned short* __restrict__ wq,
                      const unsigned short* __restrict__ wp,
                      const float* __restrict__ rpb,
                      float* __restrict__ out) {
    extern __shared__ char smem[];
    unsigned short* xb    = (unsigned short*)(smem + OFF_XB);   // x bf16 / attn-out bf16
    unsigned short* qb    = (unsigned short*)(smem + OFF_QB);   // [h][64][40] bf16 (scaled)
    unsigned short* kbuf  = (unsigned short*)(smem + OFF_KB);   // [h][64][40] bf16
    unsigned short* vt    = (unsigned short*)(smem + OFF_VT);   // [h][32][72] bf16 (transposed)
    float*          biasq = (float*)(smem + OFF_BIAS);          // [384]
    float*          attnf = (float*)(smem + OFF_ATT);           // [h][64][72] f32 logits

    const int b    = blockIdx.x;
    const int lane = threadIdx.x & 31;
    // wave index is uniform per wave -> force it into an SGPR so every tile
    // loop below is scalar-controlled (s_cbranch, not exec-mask juggling)
    const int wave = __builtin_amdgcn_readfirstlane(threadIdx.x >> 5);
    const int half = lane >> 4;
    const int lrow = lane & 15;

    const float* xwin = x + (size_t)b * NTOK * CDIM;

    // ---------------- Phase 0: warm weights, stage x (bf16, zero-pad) ------
    // prefetch the bf16 weight blocks this block will stream (128B lines)
    for (int i = threadIdx.x; i < 768; i += 256) __builtin_prefetch(wq + i * 64, 0, 0);
    if (threadIdx.x < 256)
        __builtin_prefetch(wp + threadIdx.x * 64, 0, 0);

    for (int i = threadIdx.x; i < NPAD * XB_STRIDE; i += 256) xb[i] = 0;
    for (int i = threadIdx.x; i < 3 * CDIM; i += 256) biasq[i] = qkv_b[i];
    __syncthreads();
    for (int i = threadIdx.x; i < NTOK * CDIM; i += 256) {
        int m = i >> 7, k = i & 127;                 // CDIM = 128
        xb[m * XB_STRIDE + k] = f2bf(xwin[i]);
    }
    __syncthreads();

    // ---------------- Phase 1: QKV = x @ qkv_w^T + b -----------------------
    // M=64(4 tiles), N=384(24 tiles), K=128(4 steps of 32)
    const float scale = 0.17677669529663687f;   // 32^-0.5
    for (int t = wave; t < 96; t += 8) {
        int m_tile = t & 3, n_tile = t >> 2;
        v8f acc = {};
        int arow = (m_tile * 16 + lrow) * XB_STRIDE;
        const unsigned short* bbase = wq + (n_tile * 16 + lrow) * CDIM;
        for (int kt = 0; kt < 4; ++kt) {
            Frag a, bb;
            int k0 = kt * 32 + 8 * half;
            a.q[0] = *(const uint4*)(xb + arow + k0);
            a.q[1] = *(const uint4*)(xb + arow + k0 + 16);
            int kb0 = kt * 32 + 8 * half;
            bb.q[0] = *(const uint4*)(bbase + kb0);
            bb.q[1] = *(const uint4*)(bbase + kb0 + 16);
            acc = __builtin_amdgcn_wmma_f32_16x16x32_bf16(false, a.v, false, bb.v,
                                                          (short)0, acc, false, false);
        }
        // sec/head are TILE-uniform (tiles 16-aligned, sections 128-aligned):
        // derive from n_tile so the store branch is scalar, stores straight-line.
        int sec   = n_tile >> 3;                 // 0=q 1=k 2=v
        int sub   = n_tile & 7;
        int head  = sub >> 1;
        int d     = (sub & 1) * 16 + lrow;
        float bias = biasq[n_tile * 16 + lrow];
        int mbase = m_tile * 16 + 8 * half;
        if (sec == 0) {
            for (int r = 0; r < 8; ++r)
                qb[(head * NPAD + mbase + r) * QK_STRIDE + d] = f2bf((acc[r] + bias) * scale);
        } else if (sec == 1) {
            for (int r = 0; r < 8; ++r)
                kbuf[(head * NPAD + mbase + r) * QK_STRIDE + d] = f2bf(acc[r] + bias);
        } else {
            for (int r = 0; r < 8; ++r)
                vt[(head * HD + d) * VT_STRIDE + mbase + r] = f2bf(acc[r] + bias);
        }
    }
    __syncthreads();

    // ---------------- Phase 2: logits = q @ k^T + rpb + mask ---------------
    // per head: M=64(4), N=64(4), K=32(1 WMMA)
    const float* mwin = mask + (size_t)(b & (NWIN - 1)) * NTOK * NTOK;
    for (int t = wave; t < 64; t += 8) {
        int head = t >> 4, rem = t & 15, m_tile = rem >> 2, n_tile = rem & 3;
        Frag a, bb;
        int arow = (head * NPAD + m_tile * 16 + lrow) * QK_STRIDE;
        int k0 = 8 * half;
        a.q[0] = *(const uint4*)(qb + arow + k0);
        a.q[1] = *(const uint4*)(qb + arow + k0 + 16);
        int brow = (head * NPAD + n_tile * 16 + lrow) * QK_STRIDE;
        int kb0 = 8 * half;
        bb.q[0] = *(const uint4*)(kbuf + brow + kb0);
        bb.q[1] = *(const uint4*)(kbuf + brow + kb0 + 16);
        v8f acc = {};
        acc = __builtin_amdgcn_wmma_f32_16x16x32_bf16(false, a.v, false, bb.v,
                                                      (short)0, acc, false, false);
        int n = n_tile * 16 + lrow;
        for (int r = 0; r < 8; ++r) {
            int m = m_tile * 16 + r + 8 * half;
            float val;
            if (m < NTOK && n < NTOK)
                val = acc[r] + rpb[(head * NTOK + m) * NTOK + n] + mwin[m * NTOK + n];
            else
                val = -1e30f;                      // neutralize padding
            attnf[(head * NPAD + m) * AT_STRIDE + n] = val;
        }
    }
    __syncthreads();

    // ---------------- Phase 3: softmax (1 thread per valid row) ------------
    // pass2 stores exp() back into the f32 row (in-place per index, safe);
    // pass3 scales + converts to bf16 in place (ushort@2n clobbers f32 n/2<n)
    if (threadIdx.x < HEADS * NTOK) {
        int head = threadIdx.x / NTOK, m = threadIdx.x % NTOK;
        float* rowf = attnf + (head * NPAD + m) * AT_STRIDE;
        unsigned short* rowu = (unsigned short*)rowf;
        float mx = -1e30f;
        for (int n = 0; n < NTOK; ++n) mx = fmaxf(mx, rowf[n]);
        float s = 0.0f;
        for (int n = 0; n < NTOK; ++n) {
            float e = __expf(rowf[n] - mx);
            s += e;
            rowf[n] = e;
        }
        float inv = 1.0f / s;
        for (int n = 0; n < NTOK; ++n) rowu[n] = f2bf(rowf[n] * inv);
        for (int n = NTOK; n < NPAD; ++n) rowu[n] = 0;   // zero padded keys
    }
    __syncthreads();

    // ---------------- Phase 4: attn-out = probs @ v ------------------------
    // per head: M=64(4), N=32(2), K=64(2 steps)
    const unsigned short* pa = (const unsigned short*)attnf;  // stride 144 ushort
    for (int t = wave; t < 32; t += 8) {
        int head = t >> 3, rem = t & 7, m_tile = rem >> 1, n_tile = rem & 1;
        v8f acc = {};
        int arow = (head * NPAD + m_tile * 16 + lrow) * (2 * AT_STRIDE);
        for (int kt = 0; kt < 2; ++kt) {
            Frag a, bb;
            int k0 = kt * 32 + 8 * half;
            a.q[0] = *(const uint4*)(pa + arow + k0);
            a.q[1] = *(const uint4*)(pa + arow + k0 + 16);
            int brow = (head * HD + n_tile * 16 + lrow) * VT_STRIDE;
            int kb0 = kt * 32 + 8 * half;
            bb.q[0] = *(const uint4*)(vt + brow + kb0);
            bb.q[1] = *(const uint4*)(vt + brow + kb0 + 16);
            acc = __builtin_amdgcn_wmma_f32_16x16x32_bf16(false, a.v, false, bb.v,
                                                          (short)0, acc, false, false);
        }
        int c = head * HD + n_tile * 16 + lrow;              // merged-head channel
        int mbase = m_tile * 16 + 8 * half;
        for (int r = 0; r < 8; ++r)
            xb[(mbase + r) * XB_STRIDE + c] = f2bf(acc[r]);  // reuse xb as attn-out
    }
    __syncthreads();

    // ---------------- Phase 5: out = attn_out @ proj_w^T + proj_b ----------
    // M=64(4), N=128(8), K=128(4)
    for (int t = wave; t < 32; t += 8) {
        int m_tile = t >> 3, n_tile = t & 7;
        v8f acc = {};
        int arow = (m_tile * 16 + lrow) * XB_STRIDE;
        const unsigned short* bbase = wp + (n_tile * 16 + lrow) * CDIM;
        for (int kt = 0; kt < 4; ++kt) {
            Frag a, bb;
            int k0 = kt * 32 + 8 * half;
            a.q[0] = *(const uint4*)(xb + arow + k0);
            a.q[1] = *(const uint4*)(xb + arow + k0 + 16);
            int kb0 = kt * 32 + 8 * half;
            bb.q[0] = *(const uint4*)(bbase + kb0);
            bb.q[1] = *(const uint4*)(bbase + kb0 + 16);
            acc = __builtin_amdgcn_wmma_f32_16x16x32_bf16(false, a.v, false, bb.v,
                                                          (short)0, acc, false, false);
        }
        int c = n_tile * 16 + lrow;
        float pbv = proj_b[c];
        for (int r = 0; r < 8; ++r) {
            int m = m_tile * 16 + r + 8 * half;
            if (m < NTOK)
                out[(size_t)b * NTOK * CDIM + m * CDIM + c] = acc[r] + pbv;
        }
    }
}

// ---------------------------------------------------------------------------
// Launch
// ---------------------------------------------------------------------------
extern "C" void kernel_launch(void* const* d_in, const int* in_sizes, int n_in,
                              void* d_out, int out_size, void* d_ws, size_t ws_size,
                              hipStream_t stream) {
    const float* x          = (const float*)d_in[0];
    const float* mask       = (const float*)d_in[1];
    const float* qkv_w      = (const float*)d_in[2];
    const float* qkv_b      = (const float*)d_in[3];
    const float* proj_w     = (const float*)d_in[4];
    const float* proj_b     = (const float*)d_in[5];
    const float* bias_table = (const float*)d_in[6];
    const int*   rel_idx    = (const int*)d_in[7];
    float* out = (float*)d_out;

    unsigned short* wq  = (unsigned short*)((char*)d_ws + WS_WQ);
    unsigned short* wp  = (unsigned short*)((char*)d_ws + WS_WP);
    float*          rpb = (float*)((char*)d_ws + WS_RPB);

    int prep_elems = 65536 + HEADS * NTOK * NTOK;
    swin_prep_kernel<<<(prep_elems + 255) / 256, 256, 0, stream>>>(
        qkv_w, proj_w, bias_table, rel_idx, wq, wp, rpb);

    swin_attn_kernel<<<4096, 256, LDS_BYTES, stream>>>(
        x, mask, qkv_b, proj_b, wq, wp, rpb, out);
}
